// BaseConv_47364899340393
// MI455X (gfx1250) — matrix operand order repaired and checked by reading, not compiled, not run on hardware
//
#include <hip/hip_runtime.h>
#include <hip/hip_bf16.h>
#include <stdint.h>

// ---------------- problem constants ----------------
#define NB    32          // batch
#define CIN   256
#define COUT  256
#define HH_   56
#define WW_   56
#define HP    58                      // padded spatial (1-px halo each side)
#define HW    (HH_ * WW_)             // 3136
#define NPIX  (NB * HW)               // 100352 output pixels
#define NHW   (NB * HW)
#define BLOCKC 32                     // BFP block channels
#define NBLK  (CIN / BLOCKC)          // 8

// GEMM tiling
#define MT 128                        // out-channel tile
#define NT 128                        // pixel tile
#define KSTEPS 72                     // 9 taps * 8 c-blocks

typedef _Float16 v16h __attribute__((ext_vector_type(16)));
typedef _Float16 v8h  __attribute__((ext_vector_type(8)));
typedef float    v8f  __attribute__((ext_vector_type(8)));
typedef int      v4i  __attribute__((ext_vector_type(4)));

typedef __attribute__((address_space(1))) v4i* gptr_b128;
typedef __attribute__((address_space(3))) v4i* lptr_b128;

// ---------------- CDNA5 async global->LDS copy (16B) ----------------
__device__ __forceinline__ void async_copy_b128(void* lds_dst, const void* gsrc) {
#if __has_builtin(__builtin_amdgcn_global_load_async_to_lds_b128)
    __builtin_amdgcn_global_load_async_to_lds_b128(
        (gptr_b128)(uintptr_t)gsrc,
        (lptr_b128)(unsigned)(uintptr_t)lds_dst,
        0, 0);
#else
    asm volatile("global_load_async_to_lds_b128 %0, %1, off"
                 :: "v"((unsigned)(uintptr_t)lds_dst),
                    "v"((unsigned long long)(uintptr_t)gsrc)
                 : "memory");
#endif
}

#if __has_builtin(__builtin_amdgcn_s_wait_asynccnt)
#define WAIT_ASYNC(n) __builtin_amdgcn_s_wait_asynccnt(n)
#else
#define WAIT_ASYNC(n) asm volatile("s_wait_asynccnt " #n ::: "memory")
#endif

__device__ __forceinline__ float exp2i(int e) {   // exact 2^e, e in [-126,127]
    return __uint_as_float((unsigned)((e + 127) << 23));
}

// ---------------------------------------------------------------------------
// 1) BFP quantize: x f32 NCHW -> xp f16 padded NHWC [n][h+1][w+1][c]
//    (quantized values are exactly representable in f16; halo pre-zeroed)
// ---------------------------------------------------------------------------
__global__ __launch_bounds__(256) void bfp_quant(const float* __restrict__ x,
                                                 _Float16* __restrict__ xp) {
    int t = blockIdx.x * 256 + threadIdx.x;            // < NB*NBLK*HW
    int hw = t % HW;
    int nb = (t / HW) & (NBLK - 1);
    int n  = t / (HW * NBLK);
    int h  = hw / WW_, w = hw % WW_;
    size_t ibase = ((size_t)n * CIN + (size_t)nb * BLOCKC) * HW + hw;

    float v[BLOCKC];
    float ma = 1e-12f;
    #pragma unroll
    for (int c = 0; c < BLOCKC; ++c) {
        v[c] = x[ibase + (size_t)c * HW];
        ma = fmaxf(ma, fabsf(v[c]));
    }
    int e = (int)((__float_as_uint(ma) >> 23) & 255) - 127;   // floor(log2(ma))
    e = e < -64 ? -64 : (e > 63 ? 63 : e);
    float step  = exp2i(e - 7);
    float istep = exp2i(7 - e);

    size_t obase = (((size_t)n * HP + h + 1) * HP + (w + 1)) * CIN + nb * BLOCKC;
    #pragma unroll
    for (int c = 0; c < BLOCKC; ++c) {
        float q = rintf(v[c] * istep);                 // RNE, matches jnp.round
        q = fminf(fmaxf(q, -128.0f), 127.0f);
        xp[obase + c] = (_Float16)(q * step);
    }
}

// ---------------------------------------------------------------------------
// 2) Weights OIHW f32 -> [(r*3+s)*8+cb][ko][ci] f16  (contiguous A-tile rows)
// ---------------------------------------------------------------------------
__global__ __launch_bounds__(256) void wreorder(const float* __restrict__ W,
                                                _Float16* __restrict__ wq) {
    int t = blockIdx.x * 256 + threadIdx.x;            // < 9*8*256*32
    int ci = t & 31;
    int ko = (t >> 5) & 255;
    int cb = (t >> 13) & 7;
    int rs = t >> 16;
    int r = rs / 3, s = rs % 3;
    int c = cb * BLOCKC + ci;
    wq[t] = (_Float16)W[(((size_t)ko * CIN + c) * 3 + r) * 3 + s];
}

// ---------------------------------------------------------------------------
// 3) Implicit-GEMM conv: double-buffered async LDS staging + f16 WMMA
//    block 256 thr (8 waves 4x2); wave tile 32 ko x 64 px (2x4 fragments)
// ---------------------------------------------------------------------------
__global__ __launch_bounds__(256) void conv_wmma(const _Float16* __restrict__ xp,
                                                 const _Float16* __restrict__ wq,
                                                 float* __restrict__ y) {
    __shared__ __align__(16) _Float16 As[2][MT][40];   // ko rows x 32 c (+pad)
    __shared__ __align__(16) _Float16 Bs[2][NT][40];   // px rows x 32 c (+pad)

    const int tid  = threadIdx.x;
    const int lane = tid & 31;
    const int wv   = tid >> 5;
    const int wm   = wv & 3;          // M sub-tile: wm*32
    const int wn   = wv >> 2;         // N sub-tile: wn*64
    const int ko0  = blockIdx.x * MT;
    const int p0   = blockIdx.y * NT;

    // A staging: 2 threads/row, 2x16B each
    const int a_row = tid >> 1;
    const int a_seg = (tid & 1) * 8;              // half-offset 0 / 8 (+16)
    // B staging: 2 threads/pixel, 2x16B each
    const int b_px  = tid & 127;
    const int b_seg = (tid >> 7) * 16;            // half-offset 0 / 16 (+8)
    const int bp    = p0 + b_px;
    const int b_n   = bp / HW;
    const int b_h   = (bp % HW) / WW_;
    const int b_w   = bp % WW_;

    // fragment addressing (CDNA5 wave32 WMMA layouts)
    const int mrow = lane & 15;
    const int hi   = (lane & 16) ? 1 : 0;
    const int asel = hi ? 8 : 0;      // A operand K-chunk select
    const int k0   = hi ? 8 : 0;      // B operand K-chunk select

    v8f acc[2][4] = {};

    auto stage = [&](int k, int buf) {
        const int rs = k >> 3, cb = k & 7;
        const int r = rs / 3, s = rs % 3;
        // A tile: weights, contiguous 64B rows
        const _Float16* asrc = wq + ((size_t)k * COUT + (ko0 + a_row)) * 32;
        async_copy_b128(&As[buf][a_row][a_seg],      asrc + a_seg);
        async_copy_b128(&As[buf][a_row][a_seg + 16], asrc + a_seg + 16);
        // B tile: padded-NHWC activations, no bounds checks (halo pre-zeroed;
        // +1 pad cancels the r-1/s-1 tap offset)
        const _Float16* bsrc = xp
            + (((size_t)b_n * HP + b_h + r) * HP + (b_w + s)) * CIN
            + cb * BLOCKC + b_seg;
        async_copy_b128(&Bs[buf][b_px][b_seg],     bsrc);
        async_copy_b128(&Bs[buf][b_px][b_seg + 8], bsrc + 8);
    };

    stage(0, 0);
    for (int k = 0; k < KSTEPS; ++k) {
        const int cur = k & 1;
        if (k + 1 < KSTEPS) {
            stage(k + 1, cur ^ 1);
            WAIT_ASYNC(4);            // stage-k's 4 ops done; k+1's in flight
        } else {
            WAIT_ASYNC(0);
        }
        __syncthreads();

        v16h af[2], bf[4];
        #pragma unroll
        for (int mf = 0; mf < 2; ++mf) {
            const _Float16* rowp = &As[cur][wm * 32 + mf * 16 + mrow][0];
            v8h lo  = *(const v8h*)(rowp + asel);
            v8h hi8 = *(const v8h*)(rowp + asel + 16);
            af[mf] = __builtin_shufflevector(lo, hi8,
                0,1,2,3,4,5,6,7,8,9,10,11,12,13,14,15);
        }
        #pragma unroll
        for (int nf = 0; nf < 4; ++nf) {
            const _Float16* rowp = &Bs[cur][wn * 64 + nf * 16 + mrow][0];
            v8h lo  = *(const v8h*)(rowp + k0);
            v8h hi8 = *(const v8h*)(rowp + k0 + 16);
            bf[nf] = __builtin_shufflevector(lo, hi8,
                0,1,2,3,4,5,6,7,8,9,10,11,12,13,14,15);
        }
        #pragma unroll
        for (int mf = 0; mf < 2; ++mf)
            #pragma unroll
            for (int nf = 0; nf < 4; ++nf)
                acc[mf][nf] = __builtin_amdgcn_wmma_f32_16x16x32_f16(
                    false, af[mf], false, bf[nf],
                    (short)0, acc[mf][nf], false, false);
        __syncthreads();              // done reading `cur` before k+2 overwrites
    }

    // D layout: VGPR r -> row M = r + 8*hi, col N = lane&15
    #pragma unroll
    for (int nf = 0; nf < 4; ++nf) {
        const int px = p0 + wn * 64 + nf * 16 + mrow;
        const int n  = px / HW;
        const int hw = px % HW;
        #pragma unroll
        for (int mf = 0; mf < 2; ++mf) {
            #pragma unroll
            for (int rr = 0; rr < 8; ++rr) {
                const int ko = ko0 + wm * 32 + mf * 16 + rr + 8 * hi;
                y[((size_t)n * COUT + ko) * HW + hw] = acc[mf][nf][rr];
            }
        }
    }
}

// ---------------------------------------------------------------------------
// 4) BN batch statistics: one block per channel
// ---------------------------------------------------------------------------
__global__ __launch_bounds__(256) void bn_stats(const float* __restrict__ y,
                                                float* __restrict__ sums) {
    const int c = blockIdx.x;
    float s = 0.0f, s2 = 0.0f;
    for (int i = threadIdx.x; i < NHW; i += 256) {
        int n = i / HW, hw = i % HW;
        float v = y[((size_t)n * COUT + c) * HW + hw];
        s += v; s2 += v * v;
    }
    __shared__ float sh[256], sh2[256];
    sh[threadIdx.x] = s; sh2[threadIdx.x] = s2;
    __syncthreads();
    for (int o = 128; o > 0; o >>= 1) {
        if (threadIdx.x < o) {
            sh[threadIdx.x]  += sh[threadIdx.x + o];
            sh2[threadIdx.x] += sh2[threadIdx.x + o];
        }
        __syncthreads();
    }
    if (threadIdx.x == 0) { sums[c] = sh[0]; sums[COUT + c] = sh2[0]; }
}

__global__ void bn_finalize(const float* __restrict__ sums,
                            const float* __restrict__ gamma,
                            const float* __restrict__ beta,
                            float* __restrict__ sc) {
    int c = threadIdx.x;                       // 256 threads, 1 block
    const float invn = 1.0f / (float)NHW;
    float mean = sums[c] * invn;
    float var  = sums[COUT + c] * invn - mean * mean;
    float inv  = rsqrtf(var + 1e-5f);
    float scale = gamma[c] * inv;
    sc[c]        = scale;
    sc[COUT + c] = beta[c] - mean * scale;
}

// ---------------------------------------------------------------------------
// 5) BN apply + ReLU
// ---------------------------------------------------------------------------
__global__ __launch_bounds__(256) void bn_relu(const float* __restrict__ y,
                                               const float* __restrict__ sc,
                                               float* __restrict__ out) {
    size_t i = (size_t)blockIdx.x * 256 + threadIdx.x;
    int c = (int)((i / HW) & (COUT - 1));
    out[i] = fmaxf(y[i] * sc[c] + sc[COUT + c], 0.0f);
}

// ---------------------------------------------------------------------------
extern "C" void kernel_launch(void* const* d_in, const int* in_sizes, int n_in,
                              void* d_out, int out_size, void* d_ws, size_t ws_size,
                              hipStream_t stream) {
    const float* x     = (const float*)d_in[0];
    const float* W     = (const float*)d_in[1];
    const float* gamma = (const float*)d_in[2];
    const float* beta  = (const float*)d_in[3];
    float* out = (float*)d_out;

    char* ws = (char*)d_ws;
    const size_t XP_BYTES = (size_t)NB * HP * HP * CIN * sizeof(_Float16); // 55,115,776
    const size_t WQ_BYTES = (size_t)9 * CIN * COUT * sizeof(_Float16);     //  1,179,648
    const size_t Y_BYTES  = (size_t)NB * COUT * HW * sizeof(float);        // 102,760,448
    _Float16* xp  = (_Float16*)ws;
    _Float16* wqp = (_Float16*)(ws + XP_BYTES);
    float*    y   = (float*)(ws + XP_BYTES + WQ_BYTES);
    float*    st  = (float*)(ws + XP_BYTES + WQ_BYTES + Y_BYTES);
    float*    sc  = st + 2 * COUT;

    (void)hipMemsetAsync(xp, 0, XP_BYTES, stream);     // zero halo (capturable)
    bfp_quant<<<(NB * NBLK * HW) / 256, 256, 0, stream>>>(x, xp);
    wreorder<<<(9 * CIN * COUT) / 256, 256, 0, stream>>>(W, wqp);
    conv_wmma<<<dim3(COUT / MT, NPIX / NT), 256, 0, stream>>>(xp, wqp, y);
    bn_stats<<<COUT, 256, 0, stream>>>(y, st);
    bn_finalize<<<1, COUT, 0, stream>>>(st, gamma, beta, sc);
    bn_relu<<<(int)(((size_t)NB * COUT * HW) / 256), 256, 0, stream>>>(y, sc, out);
}
